// SimpleLSTM_1769526526609
// MI455X (gfx1250) — compile-verified, matrix operand order checked by reading, not run on hardware
//
#include <hip/hip_runtime.h>

typedef __attribute__((ext_vector_type(16))) __bf16 v16bf;
typedef __attribute__((ext_vector_type(8)))  __bf16 v8bf;
typedef __attribute__((ext_vector_type(8)))  float  v8f;

namespace {
constexpr int BATCH = 64;
constexpr int SEQ   = 512;
constexpr int IN    = 512;
constexpr int HID   = 1024;
constexpr int OUT   = 512;
constexpr int COMB  = IN + HID;        // 1536
constexpr int NT_G  = (4 * HID) / 16;  // 256 column tiles across [f|i|o|c]
constexpr int KT_G  = COMB / 32;       // 48 k-steps of 32 (bf16 WMMA K)
constexpr int KT_H  = HID / 32;        // 32 k-steps in the h segment
constexpr int NT_Y  = OUT / 16;        // 32
constexpr int KT_Y  = HID / 32;        // 32
constexpr int NBLOCKS  = 64;           // persistent grid (must all be resident)
constexpr int NTHREADS = 128;          // 4 wave32 waves per block
}

// ---------------- device helpers ----------------

__device__ inline v8f zero8() {
  v8f z;
#pragma unroll
  for (int i = 0; i < 8; ++i) z[i] = 0.0f;
  return z;
}

__device__ inline v16bf cat8(v8bf a, v8bf b) {
  return __builtin_shufflevector(a, b, 0, 1, 2, 3, 4, 5, 6, 7,
                                 8, 9, 10, 11, 12, 13, 14, 15);
}

// A-operand (16x32 bf16) lane layout: lane L holds row M = L%16;
// lanes 0-15 hold K = kb+{0..7, 16..23}, lanes 16-31 hold K = kb+{8..15, 24..31}.
// p must already point at (row, kb + half8); we read p[0..7] and p[16..23].
__device__ inline v16bf loadA_bf(const __bf16* p) {
  return cat8(*(const v8bf*)p, *(const v8bf*)(p + 16));
}

__device__ inline v16bf loadA_f32(const float* p) {
  v16bf a;
#pragma unroll
  for (int j = 0; j < 8; ++j) {
    a[j]     = (__bf16)p[j];
    a[j + 8] = (__bf16)p[j + 16];
  }
  return a;
}

// B-operand from the packed weight buffer: 16 contiguous bf16 per lane per tile.
__device__ inline v16bf loadB_packed(const __bf16* base, int nT, int kT,
                                     int kTiles, int lane) {
  const __bf16* p = base + (((size_t)nT * kTiles + kT) * 32 + lane) * 16;
  return cat8(*(const v8bf*)p, *(const v8bf*)(p + 8));
}

__device__ inline v8f wmma_bf16(v16bf a, v16bf b, v8f c) {
  // (neg_a, A, neg_b, B, c_mod, C, reuse_a, reuse_b)
  return __builtin_amdgcn_wmma_f32_16x16x32_bf16(false, a, false, b,
                                                 (short)0, c, false, false);
}

__device__ inline float sigmoidf_(float v) { return 1.0f / (1.0f + __expf(-v)); }
__device__ inline float tanhf_(float v) {
  return 1.0f - 2.0f / (__expf(2.0f * v) + 1.0f);
}

// Grid-wide barrier: monotonically increasing counter (reset by init kernel).
__device__ inline void grid_sync(unsigned* cnt, unsigned target) {
  __syncthreads();
  if (threadIdx.x == 0) {
    __threadfence();
    atomicAdd(cnt, 1u);
    while (*(volatile unsigned*)cnt < target) __builtin_amdgcn_s_sleep(2);
    __threadfence();
  }
  __syncthreads();
}

// Output projection of one 16x16 tile: out[b, ts, :] = h @ W_hy + b_hy
// K loop software-pipelined by one stage; unroll 2 so the stage handoff is
// register renaming (no v_mov rotation).
__device__ inline void proj_tile(const __bf16* __restrict__ hsrc,
                                 const __bf16* Wy,
                                 float* __restrict__ out, float bhy, int ts,
                                 int pmT, int pnT, int lane, int half8,
                                 int nlane) {
  v8f acc = zero8();
  const __bf16* hrow = hsrc + (size_t)(pmT * 16 + nlane) * HID + half8;
  v16bf a = loadA_bf(hrow);
  v16bf b = loadB_packed(Wy, pnT, 0, KT_Y, lane);
#pragma unroll 2
  for (int kT = 0; kT < KT_Y - 1; ++kT) {
    v16bf an = loadA_bf(hrow + (kT + 1) * 32);
    v16bf bn = loadB_packed(Wy, pnT, kT + 1, KT_Y, lane);
    acc = wmma_bf16(a, b, acc);
    a = an; b = bn;
  }
  acc = wmma_bf16(a, b, acc);
  const int n = pnT * 16 + nlane;
#pragma unroll
  for (int r = 0; r < 8; ++r) {
    const int row = pmT * 16 + half8 + r;  // batch index
    out[((size_t)row * SEQ + ts) * OUT + n] = acc[r] + bhy;
  }
}

// ---------------- kernels ----------------

__global__ void lstm_init(const float* __restrict__ h0,
                          const float* __restrict__ c0,
                          __bf16* __restrict__ hprev,
                          float* __restrict__ c_state,
                          unsigned* __restrict__ counter) {
  const int i = blockIdx.x * blockDim.x + threadIdx.x;
  if (i == 0) *counter = 0u;
  if (i < BATCH * HID) {
    hprev[i]   = (__bf16)h0[i];
    c_state[i] = c0[i];
  }
}

// Pack [f|i|o|c] gate weights (each [1536,1024] f32 row-major) into the
// bf16 WMMA-B lane layout: dst[((nT*KT_G + kT)*32 + lane)*16 + j].
__global__ void pack_gateW(const float* __restrict__ Wf,
                           const float* __restrict__ Wi,
                           const float* __restrict__ Wo,
                           const float* __restrict__ Wc,
                           __bf16* __restrict__ dst) {
  const int id = blockIdx.x * blockDim.x + threadIdx.x;
  if (id >= NT_G * KT_G * 32) return;
  const int lane = id & 31;
  const int kT   = (id >> 5) % KT_G;
  const int nT   = (id >> 5) / KT_G;
  const int gate = nT >> 6;  // 64 column tiles per gate
  const float* W = (gate == 0) ? Wf : (gate == 1) ? Wi : (gate == 2) ? Wo : Wc;
  const int ncol = ((nT & 63) * 16) + (lane & 15);
  const int lo   = (lane >> 4) << 3;
  __bf16* p = dst + (size_t)id * 16;
#pragma unroll
  for (int j = 0; j < 8; ++j) {
    const int k0 = kT * 32 + lo + j;
    p[j]     = (__bf16)W[(size_t)k0 * HID + ncol];
    p[j + 8] = (__bf16)W[(size_t)(k0 + 16) * HID + ncol];
  }
}

// Pack W_hy [1024,512] f32 row-major into bf16 WMMA-B lane layout.
__global__ void pack_outW(const float* __restrict__ Why,
                          __bf16* __restrict__ dst) {
  const int id = blockIdx.x * blockDim.x + threadIdx.x;
  if (id >= NT_Y * KT_Y * 32) return;
  const int lane = id & 31;
  const int kT   = (id >> 5) % KT_Y;
  const int nT   = (id >> 5) / KT_Y;
  const int ncol = nT * 16 + (lane & 15);
  const int lo   = (lane >> 4) << 3;
  __bf16* p = dst + (size_t)id * 16;
#pragma unroll
  for (int j = 0; j < 8; ++j) {
    const int k0 = kT * 32 + lo + j;
    p[j]     = (__bf16)Why[(size_t)k0 * OUT + ncol];
    p[j + 8] = (__bf16)Why[(size_t)(k0 + 16) * OUT + ncol];
  }
}

__global__ void __launch_bounds__(NTHREADS, 1)
lstm_persistent(const float* __restrict__ x,
                const float* __restrict__ bF, const float* __restrict__ bI,
                const float* __restrict__ bO, const float* __restrict__ bC,
                const float* __restrict__ bHY,
                const __bf16* __restrict__ Wg,   // packed gate weights
                const __bf16* __restrict__ Wy,   // packed output weights
                __bf16* __restrict__ h0buf, __bf16* __restrict__ h1buf,
                float* __restrict__ c_state,     // d_out c_n region (live state)
                float* __restrict__ hN,          // d_out h_n region
                float* __restrict__ out,         // d_out projection region
                unsigned* __restrict__ counter) {
  const int lane  = threadIdx.x & 31;
  const int wave  = threadIdx.x >> 5;
  const int g     = blockIdx.x * (NTHREADS / 32) + wave;  // 0..255
  const int half8 = (lane >> 4) << 3;                     // 0 or 8
  const int nlane = lane & 15;

  // Gate-GEMM tile ownership: batch tile mT (0..3), H column tile jT (0..63).
  const int mT = g & 3;
  const int jT = g >> 2;
  const int mBase = mT * 16;
  const int aRow  = mBase + nlane;   // batch row this lane loads for A

  // Projection tile ownership (first 128 waves).
  const int pmT = g & 3;
  const int pnT = g >> 2;

  const int colH = jT * 16 + nlane;
  const float bf = bF[colH], bi = bI[colH], bo = bO[colH], bc = bC[colH];
  const float bhy = (g < 128) ? bHY[pnT * 16 + nlane] : 0.0f;

  unsigned phase = 0;
  const float* xrow = x + (size_t)aRow * SEQ * IN;  // + t*IN per step

  for (int t = 0; t < SEQ; ++t) {
    // Launder the weight pointers each timestep: without this, LICM proves
    // every packed-B load t-invariant, hoists ~200 tiles out of the time
    // loop, and spills them to scratch (observed in round 3: per-WMMA
    // scratch_load_b128 reloads). The empty asm makes the base opaque so B
    // stays a cheap in-loop L2 load via SGPR-base global_load.
    const __bf16* WgT = Wg;
    const __bf16* WyT = Wy;
    asm volatile("" : "+s"(WgT), "+s"(WyT));

    // ping-pong selected via ternary (keeps global address-space inference;
    // a pointer array in scratch would force FLAT loads).
    // h_{t-1} lives in buf1 for even t, buf0 for odd t.
    const __bf16* hprev = (t & 1) ? h0buf : h1buf;
    __bf16* hcur = (t & 1) ? h1buf : h0buf;

    // Warm caches for next timestep's x row (2KB = 32 lanes x 64B).
    if (t + 1 < SEQ)
      __builtin_prefetch(xrow + (size_t)(t + 1) * IN + lane * 16, 0, 1);

    // ---- fused gate GEMM: A = [h_{t-1} | x_t] (16x1536), 4 gate B tiles ----
    // Software-pipelined by one K-stage so B loads overlap the WMMAs.
    v8f aF = zero8(), aI = zero8(), aO = zero8(), aC = zero8();
    const __bf16* hrow = hprev + (size_t)aRow * HID + half8;
    const float* xr = xrow + (size_t)t * IN + half8;

    v16bf a  = loadA_bf(hrow);
    v16bf b0 = loadB_packed(WgT, 0 * 64 + jT, 0, KT_G, lane);
    v16bf b1 = loadB_packed(WgT, 1 * 64 + jT, 0, KT_G, lane);
    v16bf b2 = loadB_packed(WgT, 2 * 64 + jT, 0, KT_G, lane);
    v16bf b3 = loadB_packed(WgT, 3 * 64 + jT, 0, KT_G, lane);

    auto gemm4 = [&]() {
      aF = wmma_bf16(a, b0, aF);
      aI = wmma_bf16(a, b1, aI);
      aO = wmma_bf16(a, b2, aO);
      aC = wmma_bf16(a, b3, aC);
    };
    auto loadB4 = [&](int kT, v16bf& n0, v16bf& n1, v16bf& n2, v16bf& n3) {
      n0 = loadB_packed(WgT, 0 * 64 + jT, kT, KT_G, lane);
      n1 = loadB_packed(WgT, 1 * 64 + jT, kT, KT_G, lane);
      n2 = loadB_packed(WgT, 2 * 64 + jT, kT, KT_G, lane);
      n3 = loadB_packed(WgT, 3 * 64 + jT, kT, KT_G, lane);
    };

    // h segment: kT = 0..30 (next A also from h buffer); unroll 2 turns the
    // pipeline handoff into register renaming instead of v_mov rotation.
#pragma unroll 2
    for (int kT = 0; kT < KT_H - 1; ++kT) {
      v16bf an = loadA_bf(hrow + (kT + 1) * 32);
      v16bf n0, n1, n2, n3;
      loadB4(kT + 1, n0, n1, n2, n3);
      gemm4();
      a = an; b0 = n0; b1 = n1; b2 = n2; b3 = n3;
    }
    // handoff iteration: kT = 31, next A comes from fp32 x
    {
      v16bf an = loadA_f32(xr);
      v16bf n0, n1, n2, n3;
      loadB4(KT_H, n0, n1, n2, n3);
      gemm4();
      a = an; b0 = n0; b1 = n1; b2 = n2; b3 = n3;
    }
    // x segment: kT = 32..46 (next A from fp32 x, converted on the fly)
#pragma unroll 2
    for (int kT = KT_H; kT < KT_G - 1; ++kT) {
      v16bf an = loadA_f32(xr + (kT - KT_H + 1) * 32);
      v16bf n0, n1, n2, n3;
      loadB4(kT + 1, n0, n1, n2, n3);
      gemm4();
      a = an; b0 = n0; b1 = n1; b2 = n2; b3 = n3;
    }
    // epilogue: kT = 47
    gemm4();

    // ---- LSTM cell update for this wave's 16x16 tile of H ----
#pragma unroll
    for (int r = 0; r < 8; ++r) {
      const int row = mBase + half8 + r;        // batch index (C/D layout)
      const size_t idx = (size_t)row * HID + colH;
      const float cp = c_state[idx];
      const float fg = sigmoidf_(aF[r] + bf);
      const float ig = sigmoidf_(aI[r] + bi);
      const float og = sigmoidf_(aO[r] + bo);
      const float gg = tanhf_(aC[r] + bc);
      const float cn = fg * cp + ig * gg;
      const float hn = og * tanhf_(cn);
      c_state[idx] = cn;                        // live c state == d_out c_n
      hcur[idx]    = (__bf16)hn;
      hN[idx]      = hn;                        // last step's values survive
    }

    // ---- pipelined projection of h_{t-1} (reads the hprev buffer) ----
    if (t > 0 && g < 128)
      proj_tile(hprev, WyT, out, bhy, t - 1, pmT, pnT, lane, half8, nlane);

    grid_sync(counter, (unsigned)NBLOCKS * (++phase));
  }

  // projection of the final step's h (h_511 was written to h1buf at t=511)
  {
    const __bf16* WyT = Wy;
    asm volatile("" : "+s"(WyT));
    if (g < 128)
      proj_tile(h1buf, WyT, out, bhy, SEQ - 1, pmT, pnT, lane, half8, nlane);
  }
}

// ---------------- launcher ----------------

extern "C" void kernel_launch(void* const* d_in, const int* in_sizes, int n_in,
                              void* d_out, int out_size, void* d_ws,
                              size_t ws_size, hipStream_t stream) {
  (void)in_sizes; (void)n_in; (void)out_size; (void)ws_size;
  const float* x   = (const float*)d_in[0];
  const float* h0  = (const float*)d_in[1];
  const float* c0  = (const float*)d_in[2];
  const float* Wf  = (const float*)d_in[3];
  const float* bf_ = (const float*)d_in[4];
  const float* Wi  = (const float*)d_in[5];
  const float* bi_ = (const float*)d_in[6];
  const float* Wo  = (const float*)d_in[7];
  const float* bo_ = (const float*)d_in[8];
  const float* Wc  = (const float*)d_in[9];
  const float* bc_ = (const float*)d_in[10];
  const float* Why = (const float*)d_in[11];
  const float* bhy = (const float*)d_in[12];

  // Workspace layout (~13.9 MB): counter | packed Wg | packed Why | h ping-pong
  char* ws = (char*)d_ws;
  unsigned* counter = (unsigned*)ws;
  size_t off = 256;
  __bf16* packWg = (__bf16*)(ws + off);
  off += (size_t)NT_G * KT_G * 32 * 16 * sizeof(__bf16);  // 12,582,912 B
  __bf16* packWy = (__bf16*)(ws + off);
  off += (size_t)NT_Y * KT_Y * 32 * 16 * sizeof(__bf16);  // 1,048,576 B
  __bf16* hbuf0 = (__bf16*)(ws + off);
  off += (size_t)BATCH * HID * sizeof(__bf16);
  __bf16* hbuf1 = (__bf16*)(ws + off);

  // d_out = [output (B*S*O) | h_n (B*H) | c_n (B*H)] fp32
  float* out = (float*)d_out;
  float* hN  = out + (size_t)BATCH * SEQ * OUT;
  float* cN  = hN + (size_t)BATCH * HID;

  lstm_init<<<(BATCH * HID + 255) / 256, 256, 0, stream>>>(h0, c0, hbuf1, cN,
                                                           counter);
  pack_gateW<<<(NT_G * KT_G * 32 + 255) / 256, 256, 0, stream>>>(Wf, Wi, Wo,
                                                                 Wc, packWg);
  pack_outW<<<(NT_Y * KT_Y * 32 + 255) / 256, 256, 0, stream>>>(Why, packWy);
  lstm_persistent<<<NBLOCKS, NTHREADS, 0, stream>>>(
      x, bf_, bi_, bo_, bc_, bhy, packWg, packWy, hbuf0, hbuf1, cN, hN, out,
      counter);
}
